// PointCloudCollisionPredictor_51934744543311
// MI455X (gfx1250) — matrix operand order repaired and checked by reading, not compiled, not run on hardware
//
#include <hip/hip_runtime.h>
#include <hip/hip_bf16.h>
#include <cstdint>
#include <cstddef>

typedef _Float16 f16;
typedef _Float16 __attribute__((ext_vector_type(16))) v16h;
typedef _Float16 __attribute__((ext_vector_type(8)))  v8h;
typedef float    __attribute__((ext_vector_type(8)))  v8f;

typedef unsigned int tdm_u32x4 __attribute__((ext_vector_type(4)));
typedef int          tdm_i32x8 __attribute__((ext_vector_type(8)));
typedef int          tdm_i32x4 __attribute__((ext_vector_type(4)));

#define BN_INV 0.9999950000374997f   // 1/sqrt(1+1e-5)

// ---------------------------------------------------------------------------
// TDM staging: copy nelem f16 (contiguous) from global to LDS as one 1-D
// tensor tile. D# built per CDNA5 ISA 8.3/8.4: group0 = {count=1, lds_addr,
// global_addr[56:0], type=2}; group1 = {data_size=2B, tensor_dim0=tile_dim0=
// nelem, tensor_dim1=1, stride0=nelem}. Issued by wave 0 only (TDM ignores
// EXEC; operands are scalar). Tracked by TENSORcnt.
// This toolchain's builtin is the 6-arg form:
//   (uint32x4 g0, int32x8 g1, int32x4 g2, int32x4 g3, int32x8 pad, i32 cpol)
// ---------------------------------------------------------------------------
__device__ __forceinline__ void stage_weights(const f16* __restrict__ g,
                                              f16* __restrict__ l, unsigned int nelem)
{
#if __has_builtin(__builtin_amdgcn_tensor_load_to_lds)
  if ((threadIdx.x >> 5) == 0) {
    const unsigned long long ga = (unsigned long long)(size_t)g;
    const unsigned int la = (unsigned int)(size_t)l;   // LDS byte addr = low 32 bits
    tdm_u32x4 g0 = { 1u,                                // count=1, user-mode load
                     la,                                // lds_addr [63:32]
                     (unsigned int)ga,                  // global_addr[31:0]
                     (unsigned int)((ga >> 32) & 0x01FFFFFFu) | (2u << 30) }; // ga[56:32]|type=2
    tdm_i32x8 g1 = { (int)(1u << 16),                            // data_size=1 (2 bytes)
                     (int)((nelem & 0xFFFFu) << 16),             // tensor_dim0[15:0]
                     (int)(((nelem >> 16) & 0xFFFFu) | (1u << 16)), // td0[31:16], tensor_dim1=1
                     (int)((nelem & 0xFFFFu) << 16),             // tile_dim0 = nelem
                     0,                                           // tile_dim1/2 = 0 (unused)
                     (int)nelem,                                  // tensor_dim0_stride lo32
                     0, 0 };
    tdm_i32x4 gz4 = { 0, 0, 0, 0 };
    tdm_i32x8 gz8 = { 0, 0, 0, 0, 0, 0, 0, 0 };
    __builtin_amdgcn_tensor_load_to_lds(g0, g1, gz4, gz4, gz8, 0);
  }
#else
  for (unsigned int t = threadIdx.x; t < nelem; t += blockDim.x) l[t] = g[t];
#endif
}

__device__ __forceinline__ void stage_wait()
{
#if __has_builtin(__builtin_amdgcn_tensor_load_to_lds)
  __builtin_amdgcn_s_wait_tensorcnt(0);
#endif
}

// ---------------------------------------------------------------------------
// One 16x16 output tile: D = relu((A @ B) * (g*INV) + b), A row-major f16
// (LDS or global, stride sIn halves), B pre-transposed f16 [Ncols][kpad].
// A layout (matches B): lane<16 -> kb=0, lane>=16 -> kb=16; halves 0..15 are
// K=kb..kb+15.
// B layout: lanes 0..15 hold K=k0..k0+15, lanes 16..31 hold K=k0+16..k0+31.
// C/D layout: lane gives N=l&15; VGPR v gives M = v + (l&16 ? 8 : 0).
// ---------------------------------------------------------------------------
__device__ __forceinline__ void wmma_tile(
    const f16* __restrict__ In, int sIn,
    f16* __restrict__ Out, int sOut,
    const f16* __restrict__ WhT, int kpad,
    const float* __restrict__ gg, const float* __restrict__ bb,
    int mbase, int nbase, int K)
{
  const int l   = threadIdx.x & 31;
  const int kbA = (l < 16) ? 0 : 16;
  const int kbB = (l < 16) ? 0 : 16;
  const f16* Ap = In  + (size_t)(mbase + (l & 15)) * sIn  + kbA;
  const f16* Bp = WhT + (size_t)(nbase + (l & 15)) * kpad + kbB;
  v8f acc = {};
  for (int k0 = 0; k0 < K; k0 += 32) {
    v8h alo = *(const v8h*)(Ap + k0);
    v8h ahi = *(const v8h*)(Ap + k0 + 8);
    v16h bf = *(const v16h*)(Bp + k0);
    v16h a;
#pragma unroll
    for (int e = 0; e < 8; ++e) { a[e] = alo[e]; a[8 + e] = ahi[e]; }
    acc = __builtin_amdgcn_wmma_f32_16x16x32_f16(false, a, false, bf,
                                                 (short)0, acc, false, false);
  }
  const int   n  = nbase + (l & 15);
  const float sc = gg[n] * BN_INV;
  const float bi = bb[n];
  const int   mb = mbase + ((l & 16) ? 8 : 0);
#pragma unroll
  for (int v = 0; v < 8; ++v) {
    float x = acc[v] * sc + bi;
    Out[(size_t)(mb + v) * sOut + n] = (f16)(x > 0.f ? x : 0.f);
  }
}

// All waves of the block cooperatively cover the (M/16)x(N/16) tile grid.
__device__ __forceinline__ void mlp_layer_wmma(
    const f16* __restrict__ In, int sIn,
    f16* __restrict__ Out, int sOut,
    const f16* __restrict__ WhT, int kpad,
    const float* __restrict__ gg, const float* __restrict__ bb,
    int M, int Ncols, int K)
{
  const int wave   = threadIdx.x >> 5;
  const int nwaves = blockDim.x >> 5;
  const int mt = M >> 4, nt = Ncols >> 4;
  const int tiles = mt * nt;
  for (int t = wave; t < tiles; t += nwaves) {     // wave-uniform: EXEC stays full
    wmma_tile(In, sIn, Out, sOut, WhT, kpad, gg, bb,
              (t % mt) << 4, (t / mt) << 4, K);
  }
}

// ---------------------------------------------------------------------------
// Farthest point sampling: one block per batch, 256 threads, dists in regs.
// ---------------------------------------------------------------------------
template<int MAXP>
__global__ __launch_bounds__(256)
void fps_kernel(const float* __restrict__ xyz, int N, int S,
                int* __restrict__ fidx, float* __restrict__ newxyz)
{
  constexpr int BD = 256;
  __shared__ float sdist[BD];
  __shared__ int   sridx[BD];
  __shared__ float slast[3];
  const int b = blockIdx.x;
  const float* px = xyz + (size_t)b * N * 3;
  float dl[MAXP];
#pragma unroll
  for (int i = 0; i < MAXP; ++i) dl[i] = 1e10f;
  if (threadIdx.x == 0) fidx[(size_t)b * S] = 0;
  if (threadIdx.x < 3) {
    slast[threadIdx.x] = px[threadIdx.x];
    newxyz[(size_t)b * S * 3 + threadIdx.x] = px[threadIdx.x];
  }
  __syncthreads();
  for (int it = 1; it < S; ++it) {
    const float lx = slast[0], ly = slast[1], lz = slast[2];
    float bd = -1.f; int bi = 0;
#pragma unroll
    for (int i = 0; i < MAXP; ++i) {
      const int p = threadIdx.x + i * BD;
      if (p < N) {
        const float dx = px[p*3+0]-lx, dy = px[p*3+1]-ly, dz = px[p*3+2]-lz;
        const float d  = dx*dx + dy*dy + dz*dz;
        const float dm = dl[i] < d ? dl[i] : d;
        dl[i] = dm;
        if (dm > bd) { bd = dm; bi = p; }
      }
    }
    sdist[threadIdx.x] = bd; sridx[threadIdx.x] = bi;
    __syncthreads();
    for (int off = BD >> 1; off > 0; off >>= 1) {
      if (threadIdx.x < off) {
        const float d2 = sdist[threadIdx.x + off]; const int i2 = sridx[threadIdx.x + off];
        const float d1 = sdist[threadIdx.x];       const int i1 = sridx[threadIdx.x];
        if (d2 > d1 || (d2 == d1 && i2 < i1)) { sdist[threadIdx.x] = d2; sridx[threadIdx.x] = i2; }
      }
      __syncthreads();
    }
    const int win = sridx[0];
    if (threadIdx.x == 0) fidx[(size_t)b * S + it] = win;
    if (threadIdx.x < 3) {
      const float v = px[(size_t)win * 3 + threadIdx.x];
      slast[threadIdx.x] = v;
      newxyz[((size_t)b * S + it) * 3 + threadIdx.x] = v;
    }
    __syncthreads();
  }
}

// ---------------------------------------------------------------------------
// Ball query == first ns in-radius indices in ascending order, padded w/ first
// ---------------------------------------------------------------------------
__global__ __launch_bounds__(256)
void ball_query_kernel(const float* __restrict__ xyz, const float* __restrict__ cent,
                       int N, int S, int total, int ns, float r2,
                       int* __restrict__ outIdx)
{
  const int t = blockIdx.x * blockDim.x + threadIdx.x;
  if (t >= total) return;
  const int b = t / S;
  const float cx = cent[t*3+0], cy = cent[t*3+1], cz = cent[t*3+2];
  const float* px = xyz + (size_t)b * N * 3;
  int* op = outIdx + (size_t)t * ns;
  int cnt = 0, first = N - 1;
  bool have = false;
  for (int j = 0; j < N; ++j) {
    const float dx = px[j*3+0]-cx, dy = px[j*3+1]-cy, dz = px[j*3+2]-cz;
    if (dx*dx + dy*dy + dz*dz < r2) {
      if (!have) { have = true; first = j; }
      op[cnt++] = j;
      if (cnt == ns) break;
    }
  }
  for (; cnt < ns; ++cnt) op[cnt] = first;
}

// ---------------------------------------------------------------------------
// Fused grouped-MLP + maxpool: one block per (b,s) group. Gather -> LDS f16,
// three WMMA layers ping-ponging in LDS, column-max epilogue to global f32.
// STAGEW: weights TDM-copied into LDS (overlapped with the gather), used when
// weights + activations fit the 320KB/WGP LDS budget.
// ---------------------------------------------------------------------------
template<int NS, int KPAD, int C0, int C1, int C2, bool STAGEW>
__global__ __launch_bounds__(256)
void group_mlp_max_kernel(
    const float* __restrict__ xyz, const float* __restrict__ feats, int featC,
    int N, int S, const int* __restrict__ gidx, const float* __restrict__ cent,
    const f16* __restrict__ W0, const float* __restrict__ g0, const float* __restrict__ b0,
    const f16* __restrict__ W1, const float* __restrict__ g1, const float* __restrict__ b1,
    const f16* __restrict__ W2, const float* __restrict__ g2, const float* __restrict__ b2,
    float* __restrict__ out, int outStride, int outOffset)
{
  constexpr int WA  = (KPAD > C1) ? KPAD : C1;  // layer-0 input & layer-1 output
  constexpr int WB  = (C0 > C2) ? C0 : C2;      // layer-0 output & layer-2 output
  extern __shared__ char smem[];
  f16*  bufA = (f16*)smem;
  f16*  bufB = bufA + (size_t)NS * WA;
  int*  sidx = (int*)(bufB + (size_t)NS * WB);
  float* sc  = (float*)(sidx + NS);
  f16*  wlds = (f16*)(sc + 4);                  // weight staging region (STAGEW)

  const int blk = blockIdx.x;
  const int b = blk / S;
  const int s = blk % S;

  const f16* W0p = W0;
  const f16* W1p = W1;
  const f16* W2p = W2;
  if constexpr (STAGEW) {
    stage_weights(W0, wlds, KPAD * C0);
    stage_weights(W1, wlds + KPAD * C0, C0 * C1);
    stage_weights(W2, wlds + KPAD * C0 + C0 * C1, C1 * C2);
    W0p = wlds;
    W1p = wlds + KPAD * C0;
    W2p = wlds + KPAD * C0 + C0 * C1;
  } else {
    __builtin_prefetch(W0, 0, 3);
    __builtin_prefetch(W1, 0, 3);
    __builtin_prefetch(W2, 0, 3);
  }

  if (threadIdx.x < NS) sidx[threadIdx.x] = gidx[(size_t)blk * NS + threadIdx.x];
  if (threadIdx.x < 3)  sc[threadIdx.x]   = cent[(size_t)blk * 3 + threadIdx.x];
  __syncthreads();

  // Gather (xyz - center || feats) into bufA, zero-padded to KPAD, as f16.
  for (int t = threadIdx.x; t < NS * KPAD; t += blockDim.x) {
    const int r = t / KPAD, c = t % KPAD;
    const int j = sidx[r];
    float v = 0.f;
    if (c < 3)              v = xyz[((size_t)b * N + j) * 3 + c] - sc[c];
    else if (c - 3 < featC) v = feats[((size_t)b * N + j) * featC + (c - 3)];
    bufA[(size_t)r * WA + c] = (f16)v;
  }
  if constexpr (STAGEW) { stage_wait(); }   // TDM done before weights are read
  __syncthreads();

  mlp_layer_wmma(bufA, WA, bufB, WB, W0p, KPAD, g0, b0, NS, C0, KPAD);
  __syncthreads();
  mlp_layer_wmma(bufB, WB, bufA, WA, W1p, C0,   g1, b1, NS, C1, C0);
  __syncthreads();
  mlp_layer_wmma(bufA, WA, bufB, WB, W2p, C1,   g2, b2, NS, C2, C1);
  __syncthreads();

  float* orow = out + ((size_t)b * S + s) * outStride + outOffset;
  for (int c = threadIdx.x; c < C2; c += blockDim.x) {
    float mx = (float)bufB[c];
    for (int r = 1; r < NS; ++r) {
      const float v = (float)bufB[(size_t)r * WB + c];
      mx = v > mx ? v : mx;
    }
    orow[c] = mx;
  }
}

// ---------------------------------------------------------------------------
// Dense WMMA GEMM + BN/ReLU (SA3 per-point MLP): one wave per 16x16 tile.
// ---------------------------------------------------------------------------
__global__ __launch_bounds__(256)
void dense_mlp_kernel(const f16* __restrict__ A, int sIn,
                      const f16* __restrict__ WhT, int kpad,
                      const float* __restrict__ gg, const float* __restrict__ bb,
                      f16* __restrict__ Out, int sOut, int M, int Ncols, int K)
{
  const int nwaves = blockDim.x >> 5;
  const int wave = blockIdx.x * nwaves + (threadIdx.x >> 5);
  const int mt = M >> 4, nt = Ncols >> 4;
  if (wave >= mt * nt) return;   // wave-uniform exit
  wmma_tile(A, sIn, Out, sOut, WhT, kpad, gg, bb,
            (wave % mt) << 4, (wave / mt) << 4, K);
}

// ---------------------------------------------------------------------------
// Small utility kernels
// ---------------------------------------------------------------------------
__global__ void convert_w_kernel(const float* __restrict__ W, f16* __restrict__ WT,
                                 int cin, int cout, int kp)
{
  const int t = blockIdx.x * blockDim.x + threadIdx.x;
  if (t >= cout * kp) return;
  const int n = t / kp, k = t % kp;
  WT[t] = (k < cin) ? (f16)W[(size_t)k * cout + n] : (f16)0.f;   // transposed + K-padded
}

__global__ void concat_f3_kernel(const float* __restrict__ xyz, const float* __restrict__ feats,
                                 f16* __restrict__ out, int rows, int fc, int kp)
{
  const int t = blockIdx.x * blockDim.x + threadIdx.x;
  if (t >= rows * kp) return;
  const int r = t / kp, c = t % kp;
  float v = 0.f;
  if (c < 3)           v = xyz[(size_t)r * 3 + c];
  else if (c - 3 < fc) v = feats[(size_t)r * fc + (c - 3)];
  out[t] = (f16)v;
}

__global__ void maxpool_kernel(const f16* __restrict__ in, float* __restrict__ out,
                               int Bn, int R, int C)
{
  const int t = blockIdx.x * blockDim.x + threadIdx.x;
  if (t >= Bn * C) return;
  const int b = t / C, c = t % C;
  const f16* pp = in + (size_t)b * R * C + c;
  float mx = (float)pp[0];
  for (int r = 1; r < R; ++r) { const float v = (float)pp[(size_t)r * C]; mx = v > mx ? v : mx; }
  out[t] = mx;
}

// ---------------------------------------------------------------------------
// FC head: single block, M=16 (=batch) is exactly one WMMA M-tile.
// ---------------------------------------------------------------------------
__global__ __launch_bounds__(256)
void fc_head_kernel(const float* __restrict__ gf,
                    const f16* __restrict__ W1T, const float* __restrict__ g1, const float* __restrict__ b1,
                    const f16* __restrict__ W2T, const float* __restrict__ g2, const float* __restrict__ b2,
                    const float* __restrict__ w3, const float* __restrict__ b3,
                    float* __restrict__ out)
{
  extern __shared__ char smem[];
  f16* a0 = (f16*)smem;            // 16x1024
  f16* h1 = a0 + 16 * 1024;        // 16x512
  f16* h2 = h1 + 16 * 512;         // 16x256
  for (int t = threadIdx.x; t < 16 * 1024; t += blockDim.x) a0[t] = (f16)gf[t];
  __syncthreads();
  mlp_layer_wmma(a0, 1024, h1, 512, W1T, 1024, g1, b1, 16, 512, 1024);
  __syncthreads();
  mlp_layer_wmma(h1, 512,  h2, 256, W2T, 512,  g2, b2, 16, 256, 512);
  __syncthreads();
  if (threadIdx.x < 16) {
    float acc = 0.f;
    for (int c = 0; c < 256; ++c) acc += (float)h2[threadIdx.x * 256 + c] * w3[c];
    out[threadIdx.x] = acc + b3[0];
  }
}

// ---------------------------------------------------------------------------
// Host side
// ---------------------------------------------------------------------------
static inline int kpad32(int k) { return (k + 31) & ~31; }

struct Bump {
  char* base; size_t off;
  void* get(size_t bytes) {
    off = (off + 255) & ~(size_t)255;
    void* r = base + off; off += bytes; return r;
  }
};

static inline size_t group_lds(int NS, int KPAD, int C0, int C1, int C2, bool stageW) {
  const int WA = KPAD > C1 ? KPAD : C1;
  const int WB = C0 > C2 ? C0 : C2;
  size_t sz = (size_t)(NS * WA + NS * WB) * 2 + (size_t)NS * 4 + 64;
  if (stageW) sz += (size_t)(KPAD * C0 + C0 * C1 + C1 * C2) * 2 + 64;
  return sz;
}

extern "C" void kernel_launch(void* const* d_in, const int* in_sizes, int n_in,
                              void* d_out, int out_size, void* d_ws, size_t ws_size,
                              hipStream_t stream)
{
  (void)in_sizes; (void)n_in; (void)out_size; (void)ws_size;
  const int B = 16;

  // ---- input unpack (setup_inputs dict order, recursive) ----
  int p = 0;
  const float* pc = (const float*)d_in[p++];                      // (16,8192,3)
  static const int cin_[21]  = { 3,32,32,   3,64,64,   3,64,96,
                                 323,64,64, 323,128,128, 323,128,128,
                                 643,256,512 };
  static const int cout_[21] = { 32,32,64,  64,64,128,  64,96,128,
                                 64,64,128, 128,128,256, 128,128,256,
                                 256,512,1024 };
  const float *W_[21], *G_[21], *Bb_[21];
  for (int i = 0; i < 21; ++i) {
    W_[i]  = (const float*)d_in[p++];
    G_[i]  = (const float*)d_in[p++];
    Bb_[i] = (const float*)d_in[p++];
  }
  const float* fw1 = (const float*)d_in[p++];
  const float* fg1 = (const float*)d_in[p++];
  const float* fb1 = (const float*)d_in[p++];
  const float* fw2 = (const float*)d_in[p++];
  const float* fg2 = (const float*)d_in[p++];
  const float* fb2 = (const float*)d_in[p++];
  const float* fw3 = (const float*)d_in[p++];
  const float* fb3 = (const float*)d_in[p++];

  Bump ws{ (char*)d_ws, 0 };

  // ---- weight conversion: f32 [Cin][Cout] -> f16 transposed [Cout][Kpad] ----
  f16* WT[21];
  for (int i = 0; i < 21; ++i) {
    const int kp = kpad32(cin_[i]);
    const int tot = cout_[i] * kp;
    WT[i] = (f16*)ws.get((size_t)tot * 2);
    convert_w_kernel<<<(tot + 255) / 256, 256, 0, stream>>>(W_[i], WT[i], cin_[i], cout_[i], kp);
  }
  f16* WTf1 = (f16*)ws.get((size_t)512 * 1024 * 2);
  convert_w_kernel<<<(512 * 1024 + 255) / 256, 256, 0, stream>>>(fw1, WTf1, 1024, 512, 1024);
  f16* WTf2 = (f16*)ws.get((size_t)256 * 512 * 2);
  convert_w_kernel<<<(256 * 512 + 255) / 256, 256, 0, stream>>>(fw2, WTf2, 512, 256, 512);

  // ---- scratch buffers ----
  int*   fidx1  = (int*)  ws.get((size_t)B * 512 * 4);
  float* nx1    = (float*)ws.get((size_t)B * 512 * 3 * 4);
  float* feats1 = (float*)ws.get((size_t)B * 512 * 320 * 4);
  int*   fidx2  = (int*)  ws.get((size_t)B * 128 * 4);
  float* nx2    = (float*)ws.get((size_t)B * 128 * 3 * 4);
  float* feats2 = (float*)ws.get((size_t)B * 128 * 640 * 4);
  int*   idxbuf = (int*)  ws.get((size_t)B * 512 * 128 * 4);
  f16*   f3     = (f16*)  ws.get((size_t)B * 128 * 672 * 2);
  f16*   a1     = (f16*)  ws.get((size_t)B * 128 * 256 * 2);
  f16*   a2     = (f16*)  ws.get((size_t)B * 128 * 512 * 2);
  f16*   a3     = (f16*)  ws.get((size_t)B * 128 * 1024 * 2);
  float* gf     = (float*)ws.get((size_t)B * 1024 * 4);

  // ================= SA1 (N=8192 -> S=512) =================
  fps_kernel<32><<<B, 256, 0, stream>>>(pc, 8192, 512, fidx1, nx1);

  // branch 0: r=0.1, ns=16, 3->32->32->64, out cols [0,64)
  ball_query_kernel<<<(B * 512 + 255) / 256, 256, 0, stream>>>(pc, nx1, 8192, 512, B * 512, 16, 0.01f, idxbuf);
  group_mlp_max_kernel<16, 32, 32, 32, 64, true>
      <<<B * 512, 256, group_lds(16, 32, 32, 32, 64, true), stream>>>(
      pc, nullptr, 0, 8192, 512, idxbuf, nx1,
      WT[0], G_[0], Bb_[0], WT[1], G_[1], Bb_[1], WT[2], G_[2], Bb_[2],
      feats1, 320, 0);

  // branch 1: r=0.2, ns=32, 3->64->64->128, out cols [64,192)
  ball_query_kernel<<<(B * 512 + 255) / 256, 256, 0, stream>>>(pc, nx1, 8192, 512, B * 512, 32, 0.04f, idxbuf);
  group_mlp_max_kernel<32, 32, 64, 64, 128, true>
      <<<B * 512, 256, group_lds(32, 32, 64, 64, 128, true), stream>>>(
      pc, nullptr, 0, 8192, 512, idxbuf, nx1,
      WT[3], G_[3], Bb_[3], WT[4], G_[4], Bb_[4], WT[5], G_[5], Bb_[5],
      feats1, 320, 64);

  // branch 2: r=0.4, ns=128, 3->64->96->128, out cols [192,320)
  ball_query_kernel<<<(B * 512 + 255) / 256, 256, 0, stream>>>(pc, nx1, 8192, 512, B * 512, 128, 0.16f, idxbuf);
  group_mlp_max_kernel<128, 32, 64, 96, 128, true>
      <<<B * 512, 256, group_lds(128, 32, 64, 96, 128, true), stream>>>(
      pc, nullptr, 0, 8192, 512, idxbuf, nx1,
      WT[6], G_[6], Bb_[6], WT[7], G_[7], Bb_[7], WT[8], G_[8], Bb_[8],
      feats1, 320, 192);

  // ================= SA2 (N=512 -> S=128), feats C=320 =================
  fps_kernel<2><<<B, 256, 0, stream>>>(nx1, 512, 128, fidx2, nx2);

  // branch 0: r=0.2, ns=32, 323->64->64->128, out cols [0,128)  (weights fit LDS)
  ball_query_kernel<<<(B * 128 + 255) / 256, 256, 0, stream>>>(nx1, nx2, 512, 128, B * 128, 32, 0.04f, idxbuf);
  group_mlp_max_kernel<32, 352, 64, 64, 128, true>
      <<<B * 128, 256, group_lds(32, 352, 64, 64, 128, true), stream>>>(
      nx1, feats1, 320, 512, 128, idxbuf, nx2,
      WT[9], G_[9], Bb_[9], WT[10], G_[10], Bb_[10], WT[11], G_[11], Bb_[11],
      feats2, 640, 0);

  // branch 1: r=0.4, ns=64, 323->128->128->256, out cols [128,384) (weights too big -> L2 path)
  ball_query_kernel<<<(B * 128 + 255) / 256, 256, 0, stream>>>(nx1, nx2, 512, 128, B * 128, 64, 0.16f, idxbuf);
  group_mlp_max_kernel<64, 352, 128, 128, 256, false>
      <<<B * 128, 256, group_lds(64, 352, 128, 128, 256, false), stream>>>(
      nx1, feats1, 320, 512, 128, idxbuf, nx2,
      WT[12], G_[12], Bb_[12], WT[13], G_[13], Bb_[13], WT[14], G_[14], Bb_[14],
      feats2, 640, 128);

  // branch 2: r=0.8, ns=128, 323->128->128->256, out cols [384,640)
  ball_query_kernel<<<(B * 128 + 255) / 256, 256, 0, stream>>>(nx1, nx2, 512, 128, B * 128, 128, 0.64f, idxbuf);
  group_mlp_max_kernel<128, 352, 128, 128, 256, false>
      <<<B * 128, 256, group_lds(128, 352, 128, 128, 256, false), stream>>>(
      nx1, feats1, 320, 512, 128, idxbuf, nx2,
      WT[15], G_[15], Bb_[15], WT[16], G_[16], Bb_[16], WT[17], G_[17], Bb_[17],
      feats2, 640, 384);

  // ================= SA3: concat(xyz2, feats2) -> 643 (pad 672), dense MLP ==
  {
    const int rows = B * 128;
    concat_f3_kernel<<<(rows * 672 + 255) / 256, 256, 0, stream>>>(nx2, feats2, f3, rows, 640, 672);
    const int M = rows;
    int tiles = (M / 16) * (256 / 16);
    dense_mlp_kernel<<<(tiles + 7) / 8, 256, 0, stream>>>(f3, 672, WT[18], 672, G_[18], Bb_[18], a1, 256, M, 256, 672);
    tiles = (M / 16) * (512 / 16);
    dense_mlp_kernel<<<(tiles + 7) / 8, 256, 0, stream>>>(a1, 256, WT[19], 256, G_[19], Bb_[19], a2, 512, M, 512, 256);
    tiles = (M / 16) * (1024 / 16);
    dense_mlp_kernel<<<(tiles + 7) / 8, 256, 0, stream>>>(a2, 512, WT[20], 512, G_[20], Bb_[20], a3, 1024, M, 1024, 512);
    maxpool_kernel<<<(B * 1024 + 255) / 256, 256, 0, stream>>>(a3, gf, B, 128, 1024);
  }

  // ================= FC head =================
  {
    const size_t fcLds = (size_t)(16 * 1024 + 16 * 512 + 16 * 256) * 2;
    fc_head_kernel<<<1, 256, fcLds, stream>>>(gf, WTf1, fg1, fb1, WTf2, fg2, fb2,
                                              fw3, fb3, (float*)d_out);
  }
}
